// MultiHeadedAttention_15126874816599
// MI455X (gfx1250) — matrix operand order repaired and checked by reading, not compiled, not run on hardware
//
#include <hip/hip_runtime.h>
#include <hip/hip_bf16.h>
#include <math.h>
#include <stdint.h>

// MI455X / gfx1250: wave32, WMMA bf16 16x16x32 (f32 acc), TDM tile staging.

typedef __attribute__((ext_vector_type(16))) __bf16 v16bf;
typedef __attribute__((ext_vector_type(8)))  float  v8f;
typedef __attribute__((ext_vector_type(4)))  unsigned int v4u;
typedef __attribute__((ext_vector_type(8)))  int    v8i;
typedef __attribute__((ext_vector_type(4)))  int    v4i;

#define D_MODEL 768
#define HEADS   12
#define DK      64
#define MROWS   (512 * 128)   // 65536 projected rows

#if __has_builtin(__builtin_amdgcn_tensor_load_to_lds) && \
    __has_builtin(__builtin_amdgcn_s_wait_tensorcnt)
#define HAVE_TDM 1
#else
#define HAVE_TDM 0
#endif

// ---------- scalar helpers ----------
static __device__ __forceinline__ unsigned short f32_to_bf16(float f) {
  unsigned u = __float_as_uint(f);
  u += 0x7fffu + ((u >> 16) & 1u);          // round-to-nearest-even
  return (unsigned short)(u >> 16);
}
static __device__ __forceinline__ unsigned pack_bf16x2(float lo, float hi) {
  return (unsigned)f32_to_bf16(lo) | ((unsigned)f32_to_bf16(hi) << 16);
}
static __device__ __forceinline__ __bf16 us_bf(unsigned short x) {
  union { unsigned short u; __bf16 b; } c; c.u = x; return c.b;
}

// ---------- TDM: 2D bf16 tile load Global -> LDS (ISA 08_async_tensor §8) ----------
// pad_interval_code: DWORDs-per-row = 1<<(1+code)  (3 -> 16 dw = 32 bf16, 4 -> 32 dw = 64 bf16)
// pad_amount_code:   pad DWORDs = code+1           (1 -> 2 dw = 4 bf16, 3 -> 4 dw = 8 bf16)
#if HAVE_TDM
static __device__ __forceinline__ void tdm_load_2d_bf16(unsigned lds_off, const void* gsrc,
                                                        unsigned tile_w, unsigned tile_h,
                                                        unsigned long long row_stride_elems,
                                                        unsigned pad_interval_code,
                                                        unsigned pad_amount_code) {
  unsigned long long ga = (unsigned long long)(uintptr_t)gsrc;
  v4u g0;
  g0[0] = 1u;                                                  // count=1, user mode
  g0[1] = lds_off;                                             // lds_addr (bytes)
  g0[2] = (unsigned)(ga & 0xffffffffu);                        // global_addr[31:0]
  g0[3] = (unsigned)((ga >> 32) & 0x01ffffffu) | (2u << 30);   // addr[56:32] | type=2
  v8i g1;
  g1[0] = (int)((1u << 16)                                     // data_size = 2 bytes
              | (1u << 20)                                     // pad_enable
              | (pad_interval_code << 22)
              | (pad_amount_code << 25));
  g1[1] = (int)((tile_w & 0xffffu) << 16);                     // tensor_dim0[15:0]
  g1[2] = (int)(((tile_w >> 16) & 0xffffu) | ((tile_h & 0xffffu) << 16)); // dim0 hi | dim1 lo
  g1[3] = (int)(((tile_h >> 16) & 0xffffu) | ((tile_w & 0xffffu) << 16)); // dim1 hi | tile_dim0
  g1[4] = (int)(tile_h & 0xffffu);                             // tile_dim1 | tile_dim2=0
  g1[5] = (int)(unsigned)(row_stride_elems & 0xffffffffu);     // dim0_stride lo32
  g1[6] = (int)(unsigned)((row_stride_elems >> 32) & 0xffffu); // dim0_stride hi16
  g1[7] = 0;
  v4i z4 = {0, 0, 0, 0};
#if defined(__clang_major__) && __clang_major__ >= 23
  v8i z8 = {0, 0, 0, 0, 0, 0, 0, 0};
  __builtin_amdgcn_tensor_load_to_lds(g0, g1, z4, z4, z8, 0);
#else
  __builtin_amdgcn_tensor_load_to_lds(g0, g1, z4, z4, 0);
#endif
}
static __device__ __forceinline__ unsigned lds_offset_of(const void* p) {
  return (unsigned)(uintptr_t)p;   // generic LDS pointer: addr[31:0] = LDS byte offset
}
#endif

// ---------- WMMA fragment loaders (ISA 7.12.2, wave32) ----------
// A: 16x32 bf16 (MxK), row-major. lane 0-15: M=lane, K {0..7,16..23}; lane 16-31: K {8..15,24..31}.
static __device__ __forceinline__ v16bf load_a_bf(const unsigned short* s, int stride,
                                                  int m_off, int k_off, int lane) {
  const int half = lane >> 4;
  const int m = m_off + (lane & 15);
  v16bf a;
#pragma unroll
  for (int i = 0; i < 8; ++i) {
    int kb = k_off + ((i & 4) << 2) + (half << 3) + ((i & 3) << 1);
    a[2 * i]     = us_bf(s[m * stride + kb]);
    a[2 * i + 1] = us_bf(s[m * stride + kb + 1]);
  }
  return a;
}
// Same, fp32 source (softmax P), converted per element.
static __device__ __forceinline__ v16bf load_a_f32(const float* s, int stride,
                                                   int m_off, int k_off, int lane) {
  const int half = lane >> 4;
  const int m = m_off + (lane & 15);
  v16bf a;
#pragma unroll
  for (int i = 0; i < 8; ++i) {
    int kb = k_off + ((i & 4) << 2) + (half << 3) + ((i & 3) << 1);
    a[2 * i]     = us_bf(f32_to_bf16(s[m * stride + kb]));
    a[2 * i + 1] = us_bf(f32_to_bf16(s[m * stride + kb + 1]));
  }
  return a;
}
// B: 32x16 (KxN) from N-MAJOR LDS storage X[n][k] (i.e. B[k][n] = X[n*stride+k]).
// lane 0-15: K {0..7,16..23}, lane 16-31: K {8..15,24..31} (2/VGPR), matching the A fragment.
static __device__ __forceinline__ v16bf load_bT_bf(const unsigned short* s, int stride,
                                                   int k_off, int n_off, int lane) {
  const int half = lane >> 4;
  const int n = n_off + (lane & 15);
  v16bf b;
#pragma unroll
  for (int j = 0; j < 8; ++j) {
    int kb = k_off + ((j & 4) << 2) + (half << 3) + ((j & 3) << 1);
    b[2 * j]     = us_bf(s[n * stride + kb]);
    b[2 * j + 1] = us_bf(s[n * stride + kb + 1]);
  }
  return b;
}
static __device__ __forceinline__ v8f wmma_bf16(v16bf a, v16bf b, v8f c) {
  return __builtin_amdgcn_wmma_f32_16x16x32_bf16(false, a, false, b, (short)0, c, false, false);
}

// ---------- kernel: Wt(bf16, N-major 768x768) = transpose(convert(W)) ----------
__global__ void __launch_bounds__(256)
transpose_w_bf16(const float* __restrict__ W, unsigned short* __restrict__ Wt) {
  __shared__ unsigned short t[32][33];
  const int kb = blockIdx.x * 32;          // K base (rows of W)
  const int nb = blockIdx.y * 32;          // N base (cols of W)
  const int c = threadIdx.x & 31;
  const int r0 = threadIdx.x >> 5;         // 0..7
#pragma unroll
  for (int i = 0; i < 4; ++i) {
    int r = r0 + i * 8;
    t[r][c] = f32_to_bf16(W[(long)(kb + r) * D_MODEL + (nb + c)]);
  }
  __syncthreads();
#pragma unroll
  for (int i = 0; i < 4; ++i) {
    int r = r0 + i * 8;
    Wt[(long)(nb + r) * D_MODEL + (kb + c)] = t[c][r];
  }
}

// ---------- kernel: Msm = softmax(mask / 128) rows ----------
__global__ void mask_softmax_kernel(const float* __restrict__ mask, float* __restrict__ Msm) {
  const int r = threadIdx.x;               // 128 threads, one row each
  const float inv128 = 1.0f / 128.0f;
  float mx = -1e30f;
  for (int j = 0; j < 128; ++j) mx = fmaxf(mx, mask[r * 128 + j] * inv128);
  float s = 0.0f;
  for (int j = 0; j < 128; ++j) {
    float e = __expf(mask[r * 128 + j] * inv128 - mx);
    Msm[r * 128 + j] = e;
    s += e;
  }
  float inv = 1.0f / s;
  for (int j = 0; j < 128; ++j) Msm[r * 128 + j] *= inv;
}

// ---------- kernel: C(Mx768) = A(Mx768) @ W + bias,  W given as Wt bf16 N-major ----------
// Block tile 128x64, 8 waves * (32x32), K-step 32. A staged row-major (stride 40),
// Wt tile staged N-major (stride 40). bf16-A path uses TDM for the A tile.
template <bool A_BF16, bool OUT_F32>
__global__ void __launch_bounds__(256)
gemm768(const void* __restrict__ Ap, const unsigned short* __restrict__ Wt,
        const float* __restrict__ bias, void* __restrict__ Cp) {
  __shared__ unsigned short As[128][40];   // 128x32 bf16, stride 40 (80B rows)
  __shared__ unsigned short BsT[64][40];   // 64(N) x 32(K) bf16, stride 40

  const int tid  = threadIdx.x;
  const int lane = tid & 31;
  const int wave = tid >> 5;               // 0..7
  const int wm   = wave >> 1;              // 0..3
  const int wn   = wave & 1;               // 0..1
  const int rowBase = blockIdx.x * 128;
  const int colBase = blockIdx.y * 64;

  const float*          Af = (const float*)Ap;
  const unsigned short* Ab = (const unsigned short*)Ap;

  v8f acc[2][2];
#pragma unroll
  for (int i = 0; i < 2; ++i)
#pragma unroll
    for (int j = 0; j < 2; ++j) acc[i][j] = (v8f){0, 0, 0, 0, 0, 0, 0, 0};

  for (int kk = 0; kk < D_MODEL; kk += 32) {
#if HAVE_TDM
    if (wave == 0) {
      // Wt tile: 64 rows(N) x 32 elems(K), row stride 768 -> LDS stride 40
      tdm_load_2d_bf16(lds_offset_of(&BsT[0][0]),
                       Wt + (long)colBase * D_MODEL + kk, 32, 64, D_MODEL, 3, 3);
      if (A_BF16)
        tdm_load_2d_bf16(lds_offset_of(&As[0][0]),
                         Ab + (long)rowBase * D_MODEL + kk, 32, 128, D_MODEL, 3, 3);
    }
#else
    // Wt tile manual: 2048 elems as 256 x uint4 (8 bf16)
    {
      int r = tid >> 2, c8 = tid & 3;
      *reinterpret_cast<uint4*>(&BsT[r][c8 * 8]) =
          *reinterpret_cast<const uint4*>(Wt + (long)(colBase + r) * D_MODEL + kk + c8 * 8);
    }
    if (A_BF16) {
#pragma unroll
      for (int t = 0; t < 2; ++t) {       // 4096 elems as 512 x uint4
        int idx = tid + t * 256;
        int r = idx >> 2, c8 = idx & 3;
        *reinterpret_cast<uint4*>(&As[r][c8 * 8]) =
            *reinterpret_cast<const uint4*>(Ab + (long)(rowBase + r) * D_MODEL + kk + c8 * 8);
      }
    }
#endif
    if (!A_BF16) {
#pragma unroll
      for (int t = 0; t < 4; ++t) {       // 4096 f32 as 1024 x float4 -> packed bf16 x4
        int idx = tid + t * 256;
        int r = idx >> 3, c4 = idx & 7;
        float4 v = *reinterpret_cast<const float4*>(Af + (long)(rowBase + r) * D_MODEL + kk + c4 * 4);
        uint2 p;
        p.x = pack_bf16x2(v.x, v.y);
        p.y = pack_bf16x2(v.z, v.w);
        *reinterpret_cast<uint2*>(&As[r][c4 * 4]) = p;
      }
    }
#if HAVE_TDM
    if (wave == 0) __builtin_amdgcn_s_wait_tensorcnt(0);
#endif
    __syncthreads();

    v16bf a0 = load_a_bf(&As[0][0],  40, wm * 32,      0, lane);
    v16bf a1 = load_a_bf(&As[0][0],  40, wm * 32 + 16, 0, lane);
    v16bf b0 = load_bT_bf(&BsT[0][0], 40, 0, wn * 32,      lane);
    v16bf b1 = load_bT_bf(&BsT[0][0], 40, 0, wn * 32 + 16, lane);
    acc[0][0] = wmma_bf16(a0, b0, acc[0][0]);
    acc[0][1] = wmma_bf16(a0, b1, acc[0][1]);
    acc[1][0] = wmma_bf16(a1, b0, acc[1][0]);
    acc[1][1] = wmma_bf16(a1, b1, acc[1][1]);
    __syncthreads();
  }

  const int half = lane >> 4;
  const int n0   = lane & 15;
  float*          Cf = (float*)Cp;
  unsigned short* Cb = (unsigned short*)Cp;
#pragma unroll
  for (int i = 0; i < 2; ++i)
#pragma unroll
    for (int j = 0; j < 2; ++j)
#pragma unroll
      for (int r = 0; r < 8; ++r) {
        int gm = rowBase + wm * 32 + i * 16 + r + (half << 3);
        int gn = colBase + wn * 32 + j * 16 + n0;
        float v = acc[i][j][r] + bias[gn];
        long o = (long)gm * D_MODEL + gn;
        if (OUT_F32) Cf[o] = v; else Cb[o] = f32_to_bf16(v);
      }
}

// ---------- kernel: attention block per (x, h) ----------
// S = Qb @ Kb^T / 8 ; P = softmax(softmax(S) + Msm) ; X = P @ Vb
__global__ void __launch_bounds__(256)
attn_kernel(const unsigned short* __restrict__ Qp, const unsigned short* __restrict__ Kp,
            const unsigned short* __restrict__ Vp, const float* __restrict__ Msm,
            unsigned short* __restrict__ Xa) {
  extern __shared__ char smem_raw[];
  unsigned short* Qs = (unsigned short*)smem_raw;  // 128 x 72 bf16
  unsigned short* Ks = Qs + 128 * 72;              // 128 x 72 bf16
  unsigned short* Vs = Ks + 128 * 72;              // 128 x 72 bf16
  float*          Ss = (float*)(Vs + 128 * 72);    // 128 x 129 f32
  unsigned short* VsT = Qs;                        // reuse Q space: 64 x 136 bf16

  const int x    = blockIdx.x;                     // 0..511
  const int h    = blockIdx.y;                     // 0..11
  const int tid  = threadIdx.x;
  const int lane = tid & 31;
  const int wave = tid >> 5;                       // 0..7
  const long rowBase = (long)x * 128;
  const int  colBase = h * DK;

  // ---- stage Q/K/V 128x64 bf16 tiles (stride 72) ----
#if HAVE_TDM
  if (wave == 0) {
    tdm_load_2d_bf16(lds_offset_of(Qs), Qp + rowBase * D_MODEL + colBase, 64, 128, D_MODEL, 4, 3);
    tdm_load_2d_bf16(lds_offset_of(Ks), Kp + rowBase * D_MODEL + colBase, 64, 128, D_MODEL, 4, 3);
    tdm_load_2d_bf16(lds_offset_of(Vs), Vp + rowBase * D_MODEL + colBase, 64, 128, D_MODEL, 4, 3);
    __builtin_amdgcn_s_wait_tensorcnt(0);
  }
  __syncthreads();
#else
#pragma unroll
  for (int t = 0; t < 4; ++t) {                    // 8192 elems as 1024 x uint4 per array
    int idx = tid + t * 256;
    int r = idx >> 3, c8 = idx & 7;
    long g = (rowBase + r) * D_MODEL + (colBase + c8 * 8);
    *reinterpret_cast<uint4*>(&Qs[r * 72 + c8 * 8]) = *reinterpret_cast<const uint4*>(Qp + g);
    *reinterpret_cast<uint4*>(&Ks[r * 72 + c8 * 8]) = *reinterpret_cast<const uint4*>(Kp + g);
    *reinterpret_cast<uint4*>(&Vs[r * 72 + c8 * 8]) = *reinterpret_cast<const uint4*>(Vp + g);
  }
  __syncthreads();
#endif

  // ---- S = Q @ K^T * (1/8): each wave computes 16 rows x 128 cols ----
  {
    const int m_off = wave * 16;
    v16bf a0 = load_a_bf(Qs, 72, m_off, 0,  lane);
    v16bf a1 = load_a_bf(Qs, 72, m_off, 32, lane);
#pragma unroll
    for (int t = 0; t < 8; ++t) {
      v8f acc = {0, 0, 0, 0, 0, 0, 0, 0};
      v16bf b0 = load_bT_bf(Ks, 72, 0,  t * 16, lane);  // B[k][n] = K[n][k]
      v16bf b1 = load_bT_bf(Ks, 72, 32, t * 16, lane);
      acc = wmma_bf16(a0, b0, acc);
      acc = wmma_bf16(a1, b1, acc);
      const int half = lane >> 4;
      const int n = t * 16 + (lane & 15);
#pragma unroll
      for (int r = 0; r < 8; ++r)
        Ss[(m_off + r + (half << 3)) * 129 + n] = acc[r] * 0.125f;
    }
  }
  __syncthreads();   // Qs dead from here; safe to overwrite with VsT

  // ---- transpose V into N-major VsT[d][e] so PV B-frags vectorize ----
#pragma unroll
  for (int t = 0; t < 32; ++t) {
    int idx = tid + t * 256;                       // 0..8191
    int e = idx >> 6, d = idx & 63;
    VsT[d * 136 + e] = Vs[e * 72 + d];
  }

  // ---- dual softmax (+ mask softmax bias) per row, fp32 ----
  if (tid < 128) {
    float* row = Ss + tid * 129;
    const float* mrow = Msm + tid * 128;
    float mx = -1e30f;
#pragma unroll 4
    for (int j = 0; j < 128; ++j) mx = fmaxf(mx, row[j]);
    float s = 0.0f;
#pragma unroll 4
    for (int j = 0; j < 128; ++j) { float e = __expf(row[j] - mx); row[j] = e; s += e; }
    float inv = 1.0f / s;
    float mx2 = -1e30f;
#pragma unroll 4
    for (int j = 0; j < 128; ++j) {
      float p = row[j] * inv + mrow[j];
      row[j] = p;
      mx2 = fmaxf(mx2, p);
    }
    float s2 = 0.0f;
#pragma unroll 4
    for (int j = 0; j < 128; ++j) { float e = __expf(row[j] - mx2); row[j] = e; s2 += e; }
    float inv2 = 1.0f / s2;
#pragma unroll 4
    for (int j = 0; j < 128; ++j) row[j] *= inv2;
  }
  __syncthreads();

  // ---- X = P(128x128) @ V(128x64): each wave 16 rows x 64 cols ----
  {
    const int m_off = wave * 16;
    v16bf a[4];
#pragma unroll
    for (int ks = 0; ks < 4; ++ks) a[ks] = load_a_f32(Ss, 129, m_off, ks * 32, lane);
#pragma unroll
    for (int t = 0; t < 4; ++t) {
      v8f acc = {0, 0, 0, 0, 0, 0, 0, 0};
#pragma unroll
      for (int ks = 0; ks < 4; ++ks) {
        v16bf b = load_bT_bf(VsT, 136, ks * 32, t * 16, lane);  // B[k][n] = V[k][n]
        acc = wmma_bf16(a[ks], b, acc);
      }
      const int half = lane >> 4;
      const int n = t * 16 + (lane & 15);
#pragma unroll
      for (int r = 0; r < 8; ++r) {
        long g = (rowBase + m_off + r + (half << 3)) * D_MODEL + (colBase + n);
        Xa[g] = f32_to_bf16(acc[r]);
      }
    }
  }
}

// ---------- host ----------
extern "C" void kernel_launch(void* const* d_in, const int* in_sizes, int n_in,
                              void* d_out, int out_size, void* d_ws, size_t ws_size,
                              hipStream_t stream) {
  (void)in_sizes; (void)n_in; (void)out_size; (void)ws_size;
  const float* Q    = (const float*)d_in[0];
  const float* K    = (const float*)d_in[1];
  const float* V    = (const float*)d_in[2];
  const float* mask = (const float*)d_in[3];
  const float* Wq   = (const float*)d_in[4];
  const float* bq   = (const float*)d_in[5];
  const float* Wk   = (const float*)d_in[6];
  const float* bk   = (const float*)d_in[7];
  const float* Wv   = (const float*)d_in[8];
  const float* bv   = (const float*)d_in[9];
  const float* Wo   = (const float*)d_in[10];
  const float* bo   = (const float*)d_in[11];

  const size_t per  = (size_t)MROWS * D_MODEL;    // 50,331,648 elems
  const size_t wsz  = (size_t)D_MODEL * D_MODEL;  // 589,824 elems
  unsigned short* Qp  = (unsigned short*)d_ws;    // bf16 intermediates
  unsigned short* Kp  = Qp + per;
  unsigned short* Vp  = Kp + per;
  unsigned short* Xa  = Vp + per;
  float*          Msm = (float*)(Xa + per);       // 128x128 f32
  unsigned short* Wtq = (unsigned short*)(Msm + 128 * 128);  // bf16, transposed (N-major)
  unsigned short* Wtk = Wtq + wsz;
  unsigned short* Wtv = Wtk + wsz;
  unsigned short* Wto = Wtv + wsz;

  dim3 tgrid(D_MODEL / 32, D_MODEL / 32);         // 24 x 24
  transpose_w_bf16<<<tgrid, 256, 0, stream>>>(Wq, Wtq);
  transpose_w_bf16<<<tgrid, 256, 0, stream>>>(Wk, Wtk);
  transpose_w_bf16<<<tgrid, 256, 0, stream>>>(Wv, Wtv);
  transpose_w_bf16<<<tgrid, 256, 0, stream>>>(Wo, Wto);
  mask_softmax_kernel<<<1, 128, 0, stream>>>(mask, Msm);

  dim3 ggrid(MROWS / 128, D_MODEL / 64);          // 512 x 12
  gemm768<false, false><<<ggrid, 256, 0, stream>>>(Q, Wtq, bq, Qp);
  gemm768<false, false><<<ggrid, 256, 0, stream>>>(K, Wtk, bk, Kp);
  gemm768<false, false><<<ggrid, 256, 0, stream>>>(V, Wtv, bv, Vp);

  const size_t attn_lds = (size_t)3 * 128 * 72 * 2 + (size_t)128 * 129 * 4;  // ~120 KB
  attn_kernel<<<dim3(512, HEADS), 256, attn_lds, stream>>>(Qp, Kp, Vp, Msm, Xa);

  gemm768<true, true><<<ggrid, 256, 0, stream>>>(Xa, Wto, bo, (float*)d_out);
}
